// Decoder_25443386262174
// MI455X (gfx1250) — compile-verified, matrix operand order checked by reading, not run on hardware
//
#include <hip/hip_runtime.h>
#include <hip/hip_bf16.h>

// ---------------------------------------------------------------------------
// CDNA5 (gfx1250) pointer-network decoder.
// Precision: WMMA bf16 inputs / f32 accumulate (v_wmma_f32_16x16x32_bf16).
// K/V/P cached bf16 in workspace (L2-resident, halves per-step traffic).
// Decode: 64 persistent WGs (16 batch rows each, WMMA M=16) x 512 threads:
// waves 0-7 own the 8 WMMA N-tiles, all 16 waves share the bandwidth-bound
// GEMV/softmax phases.
// ---------------------------------------------------------------------------

typedef __attribute__((ext_vector_type(16))) __bf16        v16bf;
typedef __attribute__((ext_vector_type(8)))  float         v8f;
typedef __attribute__((ext_vector_type(8)))  unsigned int  v8u;

#define B_DIM 1024
#define S_DIM 200
#define E_DIM 128
#define H_DIM 8
#define DH    16
#define TSTEPS 199   // S-1
#define DBLK  512    // decode block size (16 waves)

__device__ __forceinline__ float bflo(unsigned u) { return __uint_as_float(u << 16); }
__device__ __forceinline__ float bfhi(unsigned u) { return __uint_as_float(u & 0xffff0000u); }
__device__ __forceinline__ float bf1(unsigned short s) { return __uint_as_float(((unsigned)s) << 16); }
__device__ __forceinline__ unsigned short f2bf(float f) {
  return (unsigned short)(__float_as_uint(f) >> 16);   // truncating bf16
}

// 32B contiguous fragment load (LDS or global) -> v16bf operand
__device__ __forceinline__ v16bf loadfrag(const unsigned short* p) {
  v8u u = *(const v8u*)p;
  return __builtin_bit_cast(v16bf, u);
}

__device__ __forceinline__ v8f wmma_bf16(v16bf a, v16bf b, v8f c) {
  // D = A(16x32 bf16) x B(32x16 bf16) + C(16x16 f32)
  return __builtin_amdgcn_wmma_f32_16x16x32_bf16(false, a, false, b, (short)0, c, false, false);
}

// ISA 7.12.2: 16-bit A 16x32 fragment element mapping.
// lane = M + 16*half ; element e <-> K via:
//   e = (rem&7) + (rem>=16 ? 8 : 0),  half = (rem>>3)&1   (rem = K % 32)
__device__ __forceinline__ void a_frag_slot(int rem, int& half, int& e) {
  half = (rem >> 3) & 1;
  e = (rem & 7) + ((rem >= 16) ? 8 : 0);
}

// ===========================================================================
// Kernel 0: swizzle a [K x 128] f32 weight into bf16 B-fragment order:
//   out[(nt*kcCount + kc)*512 + lane*16 + e]  (nt = N/16 tile, kc = K/32 tile)
// ===========================================================================
__global__ void dec_swizzle_b(const float* __restrict__ W,
                              unsigned short* __restrict__ out, int kcCount) {
  int total = 8 * kcCount * 512;
  for (int idx = blockIdx.x * blockDim.x + threadIdx.x; idx < total;
       idx += gridDim.x * blockDim.x) {
    int e    = idx & 15;
    int lane = (idx >> 4) & 31;
    int kc   = (idx >> 9) % kcCount;
    int nt   = (idx >> 9) / kcCount;
    int half = lane >> 4;
    int n    = nt * 16 + (lane & 15);
    int k    = kc * 32 + (e & 7) + (half ? 8 : 0) + ((e >= 8) ? 16 : 0);
    out[idx] = f2bf(W[k * E_DIM + n]);
  }
}

// ===========================================================================
// Kernel 1: h_hat = mean over S.  grid=B, block=E
// ===========================================================================
__global__ void dec_mean(const float* __restrict__ X, float* __restrict__ H) {
  int b = blockIdx.x, e = threadIdx.x;
  float s = 0.f;
  for (int t = 0; t < S_DIM; ++t) s += X[((size_t)b * S_DIM + t) * E_DIM + e];
  H[b * E_DIM + e] = s * (1.0f / (float)S_DIM);
}

// ===========================================================================
// Kernel 2: K/V/P = X @ {Wk,Wv,Wp}, bf16 outputs.  One block = 16 rows.
// ===========================================================================
__global__ void __launch_bounds__(256, 2)
dec_encode_gemm(const float* __restrict__ X,
                const unsigned short* __restrict__ WkF,
                const unsigned short* __restrict__ WvF,
                const unsigned short* __restrict__ WpF,
                unsigned short* __restrict__ Ko,
                unsigned short* __restrict__ Vo,
                unsigned short* __restrict__ Po) {
  __shared__ __align__(32) unsigned short sA[4 * 32 * 16];
  const int tid = threadIdx.x;
  const int m   = blockIdx.x;      // M-tile (16 rows of X)
  // Stage A fragments (f32 -> bf16, fragment order)
  for (int i = tid; i < 16 * E_DIM; i += 256) {
    int r = i >> 7, col = i & 127;
    int kc = col >> 5, rem = col & 31, half, e;
    a_frag_slot(rem, half, e);
    sA[((kc * 32) + (r + half * 16)) * 16 + e] =
        f2bf(X[((size_t)m * 16 + r) * E_DIM + col]);
  }
  __syncthreads();
  const int w = tid >> 5, lane = tid & 31;
  v16bf a[4];
#pragma unroll
  for (int kc = 0; kc < 4; ++kc) a[kc] = loadfrag(&sA[(kc * 32 + lane) * 16]);
  const unsigned short* Ws[3] = {WkF, WvF, WpF};
  unsigned short*       Os[3] = {Ko, Vo, Po};
#pragma unroll
  for (int mtx = 0; mtx < 3; ++mtx) {
    v8f acc = {};
#pragma unroll
    for (int kc = 0; kc < 4; ++kc) {
      v16bf b = loadfrag(&Ws[mtx][((w * 4 + kc) * 32 + lane) * 16]);
      acc = wmma_bf16(a[kc], b, acc);
    }
#pragma unroll
    for (int i = 0; i < 8; ++i) {   // D: M = i + 8*(lane>>4), N = lane&15
      int M = i + ((lane >> 4) << 3);
      Os[mtx][((size_t)m * 16 + M) * E_DIM + w * 16 + (lane & 15)] = f2bf(acc[i]);
    }
  }
}

// ===========================================================================
// Kernel 3: persistent decode. One block = 16 batch rows, loops t=0..198.
// 512 threads: waves 0-7 do WMMA; all waves share memory-bound phases.
// ===========================================================================
__global__ void __launch_bounds__(DBLK, 1)
dec_decode(const float* __restrict__ X, const float* __restrict__ vl,
           const float* __restrict__ vf, const float* __restrict__ Hhat,
           const unsigned short* __restrict__ Kb,
           const unsigned short* __restrict__ Vb,
           const unsigned short* __restrict__ Pb,
           const unsigned short* __restrict__ WqF,
           const unsigned short* __restrict__ WoF, float* __restrict__ out) {
  __shared__ __align__(32) unsigned short sA[12 * 32 * 16];  // h_c / ctx frags
  __shared__ float sQ[16 * E_DIM];          // q, then glimpse
  __shared__ float sS[16 * S_DIM];          // scores / attn / u
  __shared__ float sCtx[16 * E_DIM];
  __shared__ float sLast[16 * E_DIM];
  __shared__ float sFirst[16 * E_DIM];
  __shared__ unsigned char sMask[16 * S_DIM];
  __shared__ float sRed[DBLK];
  __shared__ int   sRedI[DBLK];
  __shared__ float sRowMax[16], sRowSum[16], sLogp[16];
  __shared__ int   sSel[16];

  const int tid = threadIdx.x;
  const int w = tid >> 5, lane = tid & 31;
  const int blk = blockIdx.x;               // batch rows blk*16 .. blk*16+15
  const float NEGINF = -3.3e38f;
  const size_t RAW = 0;
  const size_t LOGP = (size_t)B_DIM * TSTEPS * S_DIM;
  const size_t SOL  = LOGP + B_DIM;

  // ---- init per-row state ----
  for (int i = tid; i < 16 * E_DIM; i += DBLK) {
    int e = i & 127;
    sLast[i]  = vl[e];
    sFirst[i] = vf[e];
  }
  for (int i = tid; i < 16 * S_DIM; i += DBLK) sMask[i] = 0;
  if (tid < 16) {
    sMask[tid * S_DIM + 0] = 1;             // depot masked from start
    sLogp[tid] = 0.f;
    out[SOL + (size_t)(blk * 16 + tid) * S_DIM + 0] = 0.f;  // solution[:,0]=0
  }
  __syncthreads();

  for (int t = 0; t < TSTEPS; ++t) {
    // -- A: build h_c = [h_hat, last, first] directly in A-fragment order --
    for (int i = tid; i < 16 * 384; i += DBLK) {
      int r = i / 384, kd = i - r * 384;
      float v;
      if (kd < 128)      v = Hhat[(size_t)(blk * 16 + r) * E_DIM + kd];
      else if (kd < 256) v = sLast[r * E_DIM + kd - 128];
      else               v = sFirst[r * E_DIM + kd - 256];
      int kc = kd >> 5, rem = kd & 31, half, e;
      a_frag_slot(rem, half, e);
      sA[((kc * 32) + (r + half * 16)) * 16 + e] = f2bf(v);
    }
    __syncthreads();

    // -- B: q[16,128] = h_c[16,384] @ Wq (12 WMMA, waves 0-7) --
    if (w < 8) {
      v8f acc = {};
#pragma unroll
      for (int kc = 0; kc < 12; ++kc) {
        v16bf a = loadfrag(&sA[(kc * 32 + lane) * 16]);
        v16bf b = loadfrag(&WqF[((w * 12 + kc) * 32 + lane) * 16]);
        acc = wmma_bf16(a, b, acc);
      }
#pragma unroll
      for (int i = 0; i < 8; ++i) {
        int M = i + ((lane >> 4) << 3);
        sQ[M * E_DIM + w * 16 + (lane & 15)] = acc[i];
      }
    }
    __syncthreads();

    // -- C: per-head masked attention (per-row key/value => VALU GEMV) --
    for (int h = 0; h < H_DIM; ++h) {
      // C1: scores
      for (int i = tid; i < 16 * S_DIM; i += DBLK) {
        int r = i / S_DIM, s = i - r * S_DIM;
        const uint4* kv =
            (const uint4*)&Kb[((size_t)(blk * 16 + r) * S_DIM + s) * E_DIM + h * DH];
        const float* qp = &sQ[r * E_DIM + h * DH];
        float dot = 0.f;
#pragma unroll
        for (int c = 0; c < 2; ++c) {
          uint4  u  = kv[c];
          float4 qa = *(const float4*)(qp + c * 8);
          float4 qb = *(const float4*)(qp + c * 8 + 4);
          dot += bflo(u.x) * qa.x + bfhi(u.x) * qa.y + bflo(u.y) * qa.z +
                 bfhi(u.y) * qa.w + bflo(u.z) * qb.x + bfhi(u.z) * qb.y +
                 bflo(u.w) * qb.z + bfhi(u.w) * qb.w;
        }
        dot *= 0.25f;                                   // 1/sqrt(dh)
        sS[i] = sMask[i] ? NEGINF : dot;
      }
      __syncthreads();
      // C2: row softmax (32 threads per row, strided)
      {
        int r = tid >> 5, j = tid & 31;
        float m = NEGINF;
        for (int s = j; s < S_DIM; s += 32) m = fmaxf(m, sS[r * S_DIM + s]);
        sRed[tid] = m;
        __syncthreads();
        if (j == 0) {
          float mm = sRed[tid];
          for (int k = 1; k < 32; ++k) mm = fmaxf(mm, sRed[tid + k]);
          sRowMax[r] = mm;
        }
        __syncthreads();
        float mm = sRowMax[r], p = 0.f;
        for (int s = j; s < S_DIM; s += 32) {
          float e = __expf(sS[r * S_DIM + s] - mm);
          sS[r * S_DIM + s] = e;
          p += e;
        }
        sRed[tid] = p;
        __syncthreads();
        if (j == 0) {
          float ss = 0.f;
          for (int k = 0; k < 32; ++k) ss += sRed[tid + k];
          sRowSum[r] = ss;
        }
        __syncthreads();
      }
      // C3: ctx[r, h*16+d] = sum_s attn * V  (2 threads per (r,d), split s)
      {
        int r = tid >> 5, dd = tid & 31;
        int d = dd & 15, hp = dd >> 4;
        const unsigned short* vp =
            &Vb[(size_t)(blk * 16 + r) * S_DIM * E_DIM + h * DH + d];
        float acc = 0.f;
        int s0 = hp * 100;
        for (int s = s0; s < s0 + 100; ++s)
          acc += sS[r * S_DIM + s] * bf1(vp[(size_t)s * E_DIM]);
        sRed[tid] = acc;
        __syncthreads();
        if (dd < 16) {
          float tot = sRed[r * 32 + dd] + sRed[r * 32 + 16 + dd];
          sCtx[r * E_DIM + h * DH + dd] = tot / sRowSum[r];
        }
        __syncthreads();
      }
    }

    // -- D: glimpse = ctx @ Wo (4 WMMA, waves 0-7) --
    for (int i = tid; i < 16 * E_DIM; i += DBLK) {
      int r = i >> 7, col = i & 127;
      int kc = col >> 5, rem = col & 31, half, e;
      a_frag_slot(rem, half, e);
      sA[((kc * 32) + (r + half * 16)) * 16 + e] = f2bf(sCtx[i]);
    }
    __syncthreads();
    if (w < 8) {
      v8f acc = {};
#pragma unroll
      for (int kc = 0; kc < 4; ++kc) {
        v16bf a = loadfrag(&sA[(kc * 32 + lane) * 16]);
        v16bf b = loadfrag(&WoF[((w * 4 + kc) * 32 + lane) * 16]);
        acc = wmma_bf16(a, b, acc);
      }
#pragma unroll
      for (int i = 0; i < 8; ++i) {
        int M = i + ((lane >> 4) << 3);
        sQ[M * E_DIM + w * 16 + (lane & 15)] = acc[i];   // sQ reused: glimpse
      }
    }
    __syncthreads();

    // -- E: pointer logits u = 10*tanh(glimpse . k_ptr / sqrt(E)) --
    for (int i = tid; i < 16 * S_DIM; i += DBLK) {
      int r = i / S_DIM, s = i - r * S_DIM;
      const uint4* pp =
          (const uint4*)&Pb[((size_t)(blk * 16 + r) * S_DIM + s) * E_DIM];
      float dot = 0.f;
#pragma unroll
      for (int c = 0; c < 16; ++c) {
        uint4 u = pp[c];
        const float* gp = &sQ[r * E_DIM + c * 8];
        dot += bflo(u.x) * gp[0] + bfhi(u.x) * gp[1] + bflo(u.y) * gp[2] +
               bfhi(u.y) * gp[3] + bflo(u.z) * gp[4] + bfhi(u.z) * gp[5] +
               bflo(u.w) * gp[6] + bfhi(u.w) * gp[7];
      }
      float uu = 10.f * tanhf(dot * 0.08838834764831845f);
      out[RAW + ((size_t)(blk * 16 + r) * TSTEPS + t) * S_DIM + s] = uu;  // raw
      sS[i] = sMask[i] ? NEGINF : uu;                                     // masked
    }
    __syncthreads();

    // -- F: argmax + cross-entropy (log-softmax at argmax = -log(sum)) --
    {
      int r = tid >> 5, j = tid & 31;
      float m = NEGINF;
      int mi = 0;
      for (int s = j; s < S_DIM; s += 32) {
        float v = sS[r * S_DIM + s];
        if (v > m || (v == m && s < mi)) { m = v; mi = s; }
      }
      sRed[tid] = m; sRedI[tid] = mi;
      __syncthreads();
      if (j == 0) {
        float mm = sRed[tid]; int ii = sRedI[tid];
        for (int k = 1; k < 32; ++k) {
          float v = sRed[tid + k]; int vi = sRedI[tid + k];
          if (v > mm || (v == mm && vi < ii)) { mm = v; ii = vi; }
        }
        sRowMax[r] = mm; sSel[r] = ii;
      }
      __syncthreads();
      float mm = sRowMax[r], p = 0.f;
      for (int s = j; s < S_DIM; s += 32) p += __expf(sS[r * S_DIM + s] - mm);
      sRed[tid] = p;
      __syncthreads();
      if (j == 0) {
        float ss = 0.f;
        for (int k = 0; k < 32; ++k) ss += sRed[tid + k];
        sLogp[r] += __logf(ss);
        int ii = sSel[r];
        out[SOL + (size_t)(blk * 16 + r) * S_DIM + (t + 1)] = (float)ii;
        sMask[r * S_DIM + ii] = 1;
      }
      __syncthreads();
    }

    // -- G: state update: last = encoded[b, idx]; first set at t==0 --
    for (int i = tid; i < 16 * E_DIM; i += DBLK) {
      int r = i >> 7, e = i & 127;
      float v = X[((size_t)(blk * 16 + r) * S_DIM + sSel[r]) * E_DIM + e];
      sLast[i] = v;
      if (t == 0) sFirst[i] = v;
    }
    __syncthreads();
  }

  if (tid < 16) out[LOGP + blk * 16 + tid] = sLogp[tid];
}

// ===========================================================================
extern "C" void kernel_launch(void* const* d_in, const int* in_sizes, int n_in,
                              void* d_out, int out_size, void* d_ws, size_t ws_size,
                              hipStream_t stream) {
  (void)in_sizes; (void)n_in; (void)out_size; (void)ws_size;
  const float* X  = (const float*)d_in[0];   // encoded_inputs [B,S,E]
  const float* vl = (const float*)d_in[3];   // [E]
  const float* vf = (const float*)d_in[4];   // [E]
  const float* Wq = (const float*)d_in[5];   // [384,128]
  const float* Wk = (const float*)d_in[6];   // [128,128]
  const float* Wv = (const float*)d_in[7];
  const float* Wo = (const float*)d_in[8];
  const float* Wp = (const float*)d_in[9];
  float* out = (float*)d_out;

  // workspace layout (bytes); total ~151 MB
  char* ws = (char*)d_ws;
  const size_t ROWS = (size_t)B_DIM * S_DIM;          // 204800
  unsigned short* Kb  = (unsigned short*)(ws);
  unsigned short* Vb  = (unsigned short*)(ws + ROWS * E_DIM * 2);
  unsigned short* Pb  = (unsigned short*)(ws + 2 * ROWS * E_DIM * 2);
  float*          Hh  = (float*)(ws + 3 * ROWS * E_DIM * 2);
  char* wtab = ws + 3 * ROWS * E_DIM * 2 + (size_t)B_DIM * E_DIM * 4;
  unsigned short* WqF = (unsigned short*)(wtab);                    // 8*12*512
  unsigned short* WoF = (unsigned short*)(wtab + 49152 * 2);        // 8*4*512
  unsigned short* WkF = (unsigned short*)(wtab + (49152 + 16384) * 2);
  unsigned short* WvF = (unsigned short*)(wtab + (49152 + 2 * 16384) * 2);
  unsigned short* WpF = (unsigned short*)(wtab + (49152 + 3 * 16384) * 2);

  dec_swizzle_b<<<192, 256, 0, stream>>>(Wq, WqF, 12);
  dec_swizzle_b<<<64, 256, 0, stream>>>(Wo, WoF, 4);
  dec_swizzle_b<<<64, 256, 0, stream>>>(Wk, WkF, 4);
  dec_swizzle_b<<<64, 256, 0, stream>>>(Wv, WvF, 4);
  dec_swizzle_b<<<64, 256, 0, stream>>>(Wp, WpF, 4);
  dec_mean<<<B_DIM, E_DIM, 0, stream>>>(X, Hh);
  dec_encode_gemm<<<ROWS / 16, 256, 0, stream>>>(X, WkF, WvF, WpF, Kb, Vb, Pb);
  dec_decode<<<B_DIM / 16, DBLK, 0, stream>>>(X, vl, vf, Hh, Kb, Vb, Pb, WqF, WoF, out);
}
